// update_e_87840671137923
// MI455X (gfx1250) — matrix operand order repaired and checked by reading, not compiled, not run on hardware
//
#include <hip/hip_runtime.h>
#include <hip/hip_bf16.h>

// MI455X / gfx1250, wave32. f16 WMMA (f32 accumulate) for all three GEMMs;
// memory-bound target ~50us (1.15 GB HBM traffic @ 23.3 TB/s).

typedef __attribute__((ext_vector_type(16))) _Float16 v16h;
typedef __attribute__((ext_vector_type(8)))  _Float16 v8h;
typedef __attribute__((ext_vector_type(8)))  float    v8f;

#define N_NODES 50000
#define N_EDGES 1600000
#define HID     128
#define FIL     128
#define GAUSS   50
#define GPAD    64

// ---- WMMA fragment helpers (wave32, 16x16x32 f16) -------------------------
// B operand: weights stored [N][K] row-major f16 in LDS. Lane n=lane%16:
// halves 0..7 = K[kt*32 + hi + 0..7], halves 8..15 = K[kt*32 + 16 + hi + 0..7]
// where hi = (lane<16 ? 0 : 8).
__device__ __forceinline__ v16h ld_b_frag(const _Float16* W, int kstride,
                                          int nt, int kt, int lane) {
  int n  = nt * 16 + (lane & 15);
  int kb = kt * 32 + ((lane & 16) ? 8 : 0);
  const _Float16* p = W + n * kstride + kb;
  union { v16h v; v8h h[2]; } u;
  u.h[0] = *(const v8h*)(p);
  u.h[1] = *(const v8h*)(p + 16);
  return u.v;
}

// A operand: activations stored [M][K] row-major f16 in LDS. Lane m=lane%16:
// halves 0..7 = K[kt*32 + hi + 0..7], halves 8..15 = K[kt*32 + 16 + hi + 0..7]
// where hi = (lane<16 ? 0 : 8).
__device__ __forceinline__ v16h ld_a_frag(const _Float16* A, int kstride,
                                          int kt, int lane) {
  int m  = lane & 15;
  int hi = (lane & 16) ? 8 : 0;
  const _Float16* p = A + m * kstride + kt * 32 + hi;
  union { v16h v; v8h h[2]; } u;
  u.h[0] = *(const v8h*)(p);
  u.h[1] = *(const v8h*)(p + 16);
  return u.v;
}

__device__ __forceinline__ v8f wmma16(v16h a, v16h b, v8f c) {
  return __builtin_amdgcn_wmma_f32_16x16x32_f16(false, a, false, b,
                                                (short)0, c, false, false);
}

__device__ __forceinline__ void lds_fence() {
  __builtin_amdgcn_wave_barrier();
  asm volatile("" ::: "memory");
}

// ---- Kernel 1: vl = v @ lin_w^T  ([50000,128] x [128,128]) ---------------
// Dyn LDS: lin_w f16 (32KB) + per-wave {A f16 16x128 (4KB), O f32 16x128 (8KB)}
__global__ void __launch_bounds__(256) vl_kernel(const float* __restrict__ v,
                                                 const float* __restrict__ lin_w,
                                                 float* __restrict__ vl) {
  extern __shared__ __align__(16) char smem[];
  _Float16* sW = (_Float16*)smem;                        // [128][128]
  char* pw = smem + 128 * 128 * 2;
  const int tid = threadIdx.x, lane = tid & 31, w = tid >> 5;
  _Float16* sA = (_Float16*)(pw + w * 12288);            // [16][128] f16
  float*    sO = (float*)  (pw + w * 12288 + 4096);      // [16][128] f32

  for (int i = tid; i < 128 * 128; i += 256) sW[i] = (_Float16)lin_w[i];
  __syncthreads();

  const int tiles = (N_NODES + 15) / 16;
  int t = blockIdx.x * 8 + w;
  if (t >= tiles) return;
  const int rbase = t * 16;

  for (int i = lane; i < 16 * 128; i += 32) {
    int r = i >> 7, k = i & 127;
    int row = rbase + r;
    sA[i] = (_Float16)((row < N_NODES) ? v[(long long)row * 128 + k] : 0.f);
  }
  lds_fence();

  const int hi8 = (lane & 16) ? 8 : 0;
  const int nn  = lane & 15;
#pragma unroll
  for (int nt = 0; nt < 8; ++nt) {
    v8f acc = {};
#pragma unroll
    for (int kt = 0; kt < 4; ++kt)
      acc = wmma16(ld_a_frag(sA, 128, kt, lane),
                   ld_b_frag(sW, 128, nt, kt, lane), acc);
    int n = nt * 16 + nn;
#pragma unroll
    for (int r = 0; r < 8; ++r) sO[(r + hi8) * 128 + n] = acc[r];
  }
  lds_fence();

#pragma unroll
  for (int m = 0; m < 16; ++m) {
    int row = rbase + m;
    if (row >= N_NODES) break;
    float4 t4 = ((const float4*)(sO + m * 128))[lane];
    ((float4*)(vl + (long long)row * 128))[lane] = t4;
  }
}

// ---- Kernel 2: fused filter MLP + cutoff + gather-multiply ----------------
// Dyn LDS layout (bytes):
//   [0,16384)        sW1: mlp1_w^T as [n=128][k=64] f16 (K padded 50->64)
//   [16384,49152)    sW2: mlp2_w^T as [n=128][k=128] f16
//   [49152,49664)    sB1 f32[128]
//   [49664,50176)    sB2 f32[128]
//   [50176,+8*8192)  per-wave 8KB: A f16[16][64] then aliased as O f32[16][128]
//   [+.. ,+8*4096)   per-wave sH f16[16][128]
__global__ void __launch_bounds__(256) edge_kernel(
    const float* __restrict__ dist,
    const float* __restrict__ emb,
    const int*   __restrict__ jidx,
    const float* __restrict__ mlp1_w, const float* __restrict__ mlp1_b,
    const float* __restrict__ mlp2_w, const float* __restrict__ mlp2_b,
    const float* __restrict__ vl,
    float* __restrict__ out) {
  extern __shared__ __align__(16) char smem[];
  _Float16* sW1 = (_Float16*)smem;
  _Float16* sW2 = (_Float16*)(smem + 16384);
  float*    sB1 = (float*)(smem + 49152);
  float*    sB2 = (float*)(smem + 49664);
  char* pw = smem + 50176;
  const int tid = threadIdx.x, lane = tid & 31, w = tid >> 5;
  _Float16* sA = (_Float16*)(pw + w * 8192);                 // [16][64] f16
  float*    sO = (float*)  (pw + w * 8192);                  // [16][128] f32 (alias, A dead by then)
  _Float16* sH = (_Float16*)(pw + 8 * 8192 + w * 4096);      // [16][128] f16

  for (int i = tid; i < 128 * GPAD; i += 256) {
    int n = i >> 6, k = i & 63;
    sW1[i] = (k < GAUSS) ? (_Float16)mlp1_w[n * GAUSS + k] : (_Float16)0.f;
  }
  for (int i = tid; i < 128 * 128; i += 256) sW2[i] = (_Float16)mlp2_w[i];
  if (tid < 128) { sB1[tid] = mlp1_b[tid]; sB2[tid] = mlp2_b[tid]; }
  __syncthreads();

  const int tiles = (N_EDGES + 15) / 16;
  int t = blockIdx.x * 8 + w;
  if (t >= tiles) return;
  const long long ebase = (long long)t * 16;

  // per-edge-row metadata (rows held in lanes 0..15; 16..31 duplicate)
  int   er  = (int)ebase + (lane & 15);
  bool  vld = er < N_EDGES;
  int   j_l = vld ? jidx[er] : 0;
  float d_l = vld ? dist[er] : 0.f;
  float c_l = 0.5f * (__cosf(d_l * (3.14159265358979323846f / 10.0f)) + 1.0f);

  // stage dist_emb tile -> f16, pad K 50->64 with zeros
  for (int i = lane; i < 16 * GPAD; i += 32) {
    int r = i >> 6, k = i & 63;
    long long e = ebase + r;
    float x = (k < GAUSS && e < N_EDGES) ? emb[e * GAUSS + k] : 0.f;
    sA[i] = (_Float16)x;
  }
  lds_fence();

  const int hi8 = (lane & 16) ? 8 : 0;
  const int nn  = lane & 15;

  // mlp1: [16x64]x[64x128] -> softplus -> sH (f16, A-layout source)
#pragma unroll
  for (int nt = 0; nt < 8; ++nt) {
    v8f acc = {};
#pragma unroll
    for (int kt = 0; kt < 2; ++kt)
      acc = wmma16(ld_a_frag(sA, GPAD, kt, lane),
                   ld_b_frag(sW1, GPAD, nt, kt, lane), acc);
    int n = nt * 16 + nn;
    float bn = sB1[n];
#pragma unroll
    for (int r = 0; r < 8; ++r) {
      float x  = acc[r] + bn;
      float sp = (x > 20.f) ? x : __logf(1.f + __expf(x));
      sH[(r + hi8) * 128 + n] = (_Float16)(sp - 0.69314718055994530942f);
    }
  }
  lds_fence();

  // mlp2: [16x128]x[128x128] -> +b2 -> sO (f32; aliases dead A region)
#pragma unroll
  for (int nt = 0; nt < 8; ++nt) {
    v8f acc = {};
#pragma unroll
    for (int kt = 0; kt < 4; ++kt)
      acc = wmma16(ld_a_frag(sH, 128, kt, lane),
                   ld_b_frag(sW2, 128, nt, kt, lane), acc);
    int n = nt * 16 + nn;
    float bn = sB2[n];
#pragma unroll
    for (int r = 0; r < 8; ++r) sO[(r + hi8) * 128 + n] = acc[r] + bn;
  }
  lds_fence();

  // final: out[e,:] = sO[m,:] * C[m] * vl[j[m],:]  (fully coalesced float4)
#pragma unroll
  for (int m = 0; m < 16; ++m) {
    long long e = ebase + m;
    if (e >= N_EDGES) break;
    int   jm = __shfl(j_l, m);
    float cm = __shfl(c_l, m);
    float4 a4 = ((const float4*)(vl + (long long)jm * 128))[lane];
    float4 t4 = ((const float4*)(sO + m * 128))[lane];
    float4 r4;
    r4.x = t4.x * cm * a4.x;
    r4.y = t4.y * cm * a4.y;
    r4.z = t4.z * cm * a4.z;
    r4.w = t4.w * cm * a4.w;
    ((float4*)(out + e * 128))[lane] = r4;
  }
}

extern "C" void kernel_launch(void* const* d_in, const int* in_sizes, int n_in,
                              void* d_out, int out_size, void* d_ws, size_t ws_size,
                              hipStream_t stream) {
  const float* v      = (const float*)d_in[0];
  const float* dist   = (const float*)d_in[1];
  const float* emb    = (const float*)d_in[2];
  const int*   eidx   = (const int*)d_in[3];   // edge_index[0] = first E entries
  const float* lin_w  = (const float*)d_in[4];
  const float* mlp1_w = (const float*)d_in[5];
  const float* mlp1_b = (const float*)d_in[6];
  const float* mlp2_w = (const float*)d_in[7];
  const float* mlp2_b = (const float*)d_in[8];
  float* out = (float*)d_out;
  float* vl  = (float*)d_ws;                   // 50000*128 f32 = 25.6 MB

  const int vl_smem   = 32768 + 8 * 12288;     // 131072 B
  const int edge_smem = 50176 + 8 * 8192 + 8 * 4096;  // 148480 B
  hipFuncSetAttribute((const void*)vl_kernel,
                      hipFuncAttributeMaxDynamicSharedMemorySize, vl_smem);
  hipFuncSetAttribute((const void*)edge_kernel,
                      hipFuncAttributeMaxDynamicSharedMemorySize, edge_smem);

  int vtiles = (N_NODES + 15) / 16;            // 3125
  int vblocks = (vtiles + 7) / 8;              // 391
  vl_kernel<<<vblocks, 256, vl_smem, stream>>>(v, lin_w, vl);

  int etiles = (N_EDGES + 15) / 16;            // 100000
  int eblocks = (etiles + 7) / 8;              // 12500
  edge_kernel<<<eblocks, 256, edge_smem, stream>>>(dist, emb, eidx,
                                                   mlp1_w, mlp1_b,
                                                   mlp2_w, mlp2_b,
                                                   vl, out);
}
